// CodeGenAttention_55327768708041
// MI455X (gfx1250) — matrix operand order repaired and checked by reading, not compiled, not run on hardware
//
#include <hip/hip_runtime.h>
#include <hip/hip_bf16.h>

// ---------------------------------------------------------------------------
// CodeGen (GPT-J style) attention for MI455X / gfx1250, wave32 + WMMA bf16.
// B=2, S=2048, D=4096, H=16, head_dim=256, rotary_dim=64, MP=4.
// ---------------------------------------------------------------------------

typedef __attribute__((ext_vector_type(16))) __bf16 v16bf;
typedef __attribute__((ext_vector_type(8)))  float  v8f;

// Exact pointee type from hipcc's diagnostic for the async-DMA builtin:
typedef int v4i_async __attribute__((vector_size(4 * sizeof(int))));

#if __has_builtin(__builtin_amdgcn_global_load_async_to_lds_b128) && \
    __has_builtin(__builtin_amdgcn_s_wait_asynccnt)
#define USE_ASYNC 1
#define ASYNC_B128(gp, lp, off)                                        \
  __builtin_amdgcn_global_load_async_to_lds_b128(                      \
      (v4i_async*)(void*)(gp), (v4i_async*)(void*)(lp), (off), 0)
#else
#define USE_ASYNC 0
#endif

union FragU { uint4 u[2]; v16bf v; };

// A-matrix fragment (16x32 bf16, MxK) per CDNA5 ISA layout:
//   lanes 0-15 : row m, K[0..7] then K[16..23]
//   lanes 16-31: row m, K[8..15] then K[24..31]
static __device__ __forceinline__ v16bf frag_A(const __bf16* tile, int row0,
                                               int stride, int k0) {
  int lane = threadIdx.x & 31;
  int hf = lane >> 4;
  int r  = lane & 15;
  const __bf16* p = tile + (size_t)(row0 + r) * stride + k0 + hf * 8;
  FragU f;
  f.u[0] = *(const uint4*)p;
  f.u[1] = *(const uint4*)(p + 16);
  return f.v;
}

// B-matrix fragment (32x16 bf16, KxN) from an N-major (transposed) tile:
//   lanes 0-15 : col n, K[0..7] then K[16..23]; lanes 16-31: col n, K[8..15] then K[24..31]
static __device__ __forceinline__ v16bf frag_B(const __bf16* tileT, int col0,
                                               int stride, int k0) {
  int lane = threadIdx.x & 31;
  int hf = lane >> 4;
  int r  = lane & 15;
  const __bf16* p = tileT + (size_t)(col0 + r) * stride + k0 + hf * 8;
  FragU f;
  f.u[0] = *(const uint4*)p;
  f.u[1] = *(const uint4*)(p + 16);
  return f.v;
}

static __device__ __forceinline__ v8f wmma_bf16(v16bf a, v16bf b, v8f c) {
  return __builtin_amdgcn_wmma_f32_16x16x32_bf16(false, a, false, b,
                                                 (short)0, c, false, false);
}

// ---------------------------------------------------------------------------
// fp32 -> bf16 conversion (grid-stride), row-major preserved
// ---------------------------------------------------------------------------
__global__ void cvt_f32_bf16(const float* __restrict__ src,
                             __bf16* __restrict__ dst, size_t n) {
  size_t i = (size_t)blockIdx.x * blockDim.x + threadIdx.x;
  size_t stride = (size_t)gridDim.x * blockDim.x;
  for (; i < n; i += stride) dst[i] = (__bf16)src[i];
}

// ---------------------------------------------------------------------------
// fp32 -> bf16 convert + transpose: src[R][C] f32 -> dst[C][R] bf16.
// 64x64 tiles through LDS so both global streams are coalesced.
// ---------------------------------------------------------------------------
__global__ __launch_bounds__(256) void cvt_transpose_f32_bf16(
    const float* __restrict__ src, __bf16* __restrict__ dst, int R, int C) {
  __shared__ __bf16 t[64][80];  // pad to 160B rows (16B-aligned, conflict-free)
  int r0 = blockIdx.y * 64, c0 = blockIdx.x * 64;
  int tr = threadIdx.x >> 2;        // 0..63
  int cs = (threadIdx.x & 3) * 16;  // 0,16,32,48

  const float* sp = src + (size_t)(r0 + tr) * C + c0 + cs;
#pragma unroll
  for (int v = 0; v < 4; ++v) {
    float4 f = *(const float4*)(sp + v * 4);
    t[tr][cs + v * 4 + 0] = (__bf16)f.x;
    t[tr][cs + v * 4 + 1] = (__bf16)f.y;
    t[tr][cs + v * 4 + 2] = (__bf16)f.z;
    t[tr][cs + v * 4 + 3] = (__bf16)f.w;
  }
  __syncthreads();

  int dc = tr;   // local output row (= source column)
  int rs = cs;   // local output col (= source row)
  union { uint4 u[2]; __bf16 h[16]; } o;
#pragma unroll
  for (int e = 0; e < 16; ++e) o.h[e] = t[rs + e][dc];
  __bf16* dp = dst + (size_t)(c0 + dc) * R + r0 + rs;
  *(uint4*)dp       = o.u[0];
  *(uint4*)(dp + 8) = o.u[1];
}

// ---------------------------------------------------------------------------
// bf16 transpose per batch slice: src[z][R][C] -> dst[z][C][R], 64x64 tiles.
// Used once to make V d-major for the PV WMMA B-operand.
// ---------------------------------------------------------------------------
__global__ __launch_bounds__(256) void transpose_bf16(
    const __bf16* __restrict__ src, __bf16* __restrict__ dst, int R, int C) {
  __shared__ __bf16 t[64][80];
  size_t zoff = (size_t)blockIdx.z * R * C;
  int r0 = blockIdx.y * 64, c0 = blockIdx.x * 64;
  int tr = threadIdx.x >> 2;
  int cs = (threadIdx.x & 3) * 16;

  const __bf16* sp = src + zoff + (size_t)(r0 + tr) * C + c0 + cs;
  *(uint4*)(&t[tr][cs])     = *(const uint4*)sp;
  *(uint4*)(&t[tr][cs + 8]) = *(const uint4*)(sp + 8);
  __syncthreads();

  int dc = tr, rs = cs;
  union { uint4 u[2]; __bf16 h[16]; } o;
#pragma unroll
  for (int e = 0; e < 16; ++e) o.h[e] = t[rs + e][dc];
  __bf16* dp = dst + zoff + (size_t)(c0 + dc) * R + r0 + rs;
  *(uint4*)dp       = o.u[0];
  *(uint4*)(dp + 8) = o.u[1];
}

// ---------------------------------------------------------------------------
// GEMM: C[M,N] (fp32) = A[M,K] (bf16 row-major) * Bt[N,K] (bf16 N-major).
// Block tile 128x128, K-step 32, 256 threads = 8 waves, wave tile 64x32.
// Async-DMA double-buffered LDS when available.
// ---------------------------------------------------------------------------
#define GT_M 128
#define GT_N 128
#define GT_K 32

template <int N, int K>
__global__ __launch_bounds__(256) void gemm_bf16_f32(
    const __bf16* __restrict__ A, const __bf16* __restrict__ Bt,
    float* __restrict__ C) {
#if USE_ASYNC
  __shared__ __bf16 As[2][GT_M * GT_K];
  __shared__ __bf16 Bts[2][GT_N * GT_K];
#else
  __shared__ __bf16 As[1][GT_M * GT_K];
  __shared__ __bf16 Bts[1][GT_N * GT_K];
#endif

  int tid  = threadIdx.x;
  int lane = tid & 31;
  int w    = tid >> 5;
  int hf   = lane >> 4;
  int r    = lane & 15;

  int bm = blockIdx.y * GT_M;
  int bn = blockIdx.x * GT_N;
  int wm = (w >> 2) * 64;  // 0 or 64
  int wn = (w & 3) * 32;   // 0,32,64,96

  v8f zero = {0.f, 0.f, 0.f, 0.f, 0.f, 0.f, 0.f, 0.f};
  v8f acc[4][2];
#pragma unroll
  for (int i = 0; i < 4; ++i)
#pragma unroll
    for (int j = 0; j < 2; ++j) acc[i][j] = zero;

  int arow = tid >> 1, acs = (tid & 1) * 16;  // each thread stages 16 elems
  const __bf16* gA = A + (size_t)(bm + arow) * K + acs;
  const __bf16* gB = Bt + (size_t)(bn + arow) * K + acs;
  constexpr int NK = K / GT_K;

#if USE_ASYNC
  // prologue: DMA tile 0 into buffer 0
  ASYNC_B128(gA, &As[0][arow * GT_K + acs], 0);
  ASYNC_B128(gA, &As[0][arow * GT_K + acs], 16);
  ASYNC_B128(gB, &Bts[0][arow * GT_K + acs], 0);
  ASYNC_B128(gB, &Bts[0][arow * GT_K + acs], 16);
  for (int kk = 0; kk < NK; ++kk) {
    int cur = kk & 1;
    if (kk + 1 < NK) {
      const __bf16* ga = gA + (kk + 1) * GT_K;
      const __bf16* gb = gB + (kk + 1) * GT_K;
      ASYNC_B128(ga, &As[1 - cur][arow * GT_K + acs], 0);
      ASYNC_B128(ga, &As[1 - cur][arow * GT_K + acs], 16);
      ASYNC_B128(gb, &Bts[1 - cur][arow * GT_K + acs], 0);
      ASYNC_B128(gb, &Bts[1 - cur][arow * GT_K + acs], 16);
      __builtin_amdgcn_s_wait_asynccnt(4);  // tile kk landed (in-order)
    } else {
      __builtin_amdgcn_s_wait_asynccnt(0);
    }
    __syncthreads();
    v16bf bfr[2];
#pragma unroll
    for (int j = 0; j < 2; ++j) bfr[j] = frag_B(Bts[cur], wn + j * 16, GT_K, 0);
#pragma unroll
    for (int i = 0; i < 4; ++i) {
      v16bf af = frag_A(As[cur], wm + i * 16, GT_K, 0);
#pragma unroll
      for (int j = 0; j < 2; ++j) acc[i][j] = wmma_bf16(af, bfr[j], acc[i][j]);
    }
    __syncthreads();
  }
#else
  for (int kk = 0; kk < NK; ++kk) {
    const __bf16* ga = gA + kk * GT_K;
    const __bf16* gb = gB + kk * GT_K;
    uint4 a0 = *(const uint4*)ga, a1 = *(const uint4*)(ga + 8);
    uint4 b0 = *(const uint4*)gb, b1 = *(const uint4*)(gb + 8);
    *(uint4*)(&As[0][arow * GT_K + acs])      = a0;
    *(uint4*)(&As[0][arow * GT_K + acs + 8])  = a1;
    *(uint4*)(&Bts[0][arow * GT_K + acs])     = b0;
    *(uint4*)(&Bts[0][arow * GT_K + acs + 8]) = b1;
    __syncthreads();
    v16bf bfr[2];
#pragma unroll
    for (int j = 0; j < 2; ++j) bfr[j] = frag_B(Bts[0], wn + j * 16, GT_K, 0);
#pragma unroll
    for (int i = 0; i < 4; ++i) {
      v16bf af = frag_A(As[0], wm + i * 16, GT_K, 0);
#pragma unroll
      for (int j = 0; j < 2; ++j) acc[i][j] = wmma_bf16(af, bfr[j], acc[i][j]);
    }
    __syncthreads();
  }
#endif

  // epilogue: C layout — VGPR rr: row = rr + 8*hf, col = lane&15
  float* cbase = C + (size_t)(bm + wm + hf * 8) * N + bn + wn + r;
#pragma unroll
  for (int i = 0; i < 4; ++i)
#pragma unroll
    for (int j = 0; j < 2; ++j)
#pragma unroll
      for (int rr = 0; rr < 8; ++rr)
        cbase[(size_t)(i * 16 + rr) * N + j * 16] = acc[i][j][rr];
}

// ---------------------------------------------------------------------------
// Rotary + q/v/k scatter.  qkv layout: [B,S,MP=4, 3*1024] with q|v|k order.
// Writes Q (pre-scaled by 1/16), K, V as bf16 [B,H,S,256].
// ---------------------------------------------------------------------------
__global__ __launch_bounds__(128) void rotary_scatter(
    const float* __restrict__ qkv, const int* __restrict__ pos_ids,
    __bf16* __restrict__ Qb, __bf16* __restrict__ Kb, __bf16* __restrict__ Vb) {
  int h = blockIdx.x;   // 0..15
  int s = blockIdx.y;   // 0..2047
  int b = blockIdx.z;   // 0..1
  int p = threadIdx.x;  // 0..127 (dim pair)
  int g = h >> 2, hg = h & 3;

  size_t rowoff = ((size_t)(b * 2048 + s)) * 12288 + (size_t)g * 3072 + hg * 256;
  const float* qp = qkv + rowoff;
  const float* vp = qkv + rowoff + 1024;
  const float* kp = qkv + rowoff + 2048;

  int d0 = 2 * p, d1 = 2 * p + 1;
  float q0 = qp[d0], q1 = qp[d1];
  float k0 = kp[d0], k1 = kp[d1];
  float v0 = vp[d0], v1 = vp[d1];

  if (d0 < 64) {
    float pos = (float)pos_ids[b * 2048 + s];
    float inv = __powf(10000.f, -((float)d0) / 64.f);
    float ang = pos * inv;
    float sn, cs;
    __sincosf(ang, &sn, &cs);
    float nq0 = q0 * cs - q1 * sn, nq1 = q1 * cs + q0 * sn;
    float nk0 = k0 * cs - k1 * sn, nk1 = k1 * cs + k0 * sn;
    q0 = nq0; q1 = nq1; k0 = nk0; k1 = nk1;
  }
  const float qs = 1.0f / 16.0f;  // 1/sqrt(head_dim)
  size_t o = (((size_t)(b * 16 + h)) * 2048 + s) * 256;
  Qb[o + d0] = (__bf16)(q0 * qs); Qb[o + d1] = (__bf16)(q1 * qs);
  Kb[o + d0] = (__bf16)k0;        Kb[o + d1] = (__bf16)k1;
  Vb[o + d0] = (__bf16)v0;        Vb[o + d1] = (__bf16)v1;
}

// ---------------------------------------------------------------------------
// Flash attention: one block per (b, h, 64-query tile). 256 threads = 8 waves.
// QK^T and PV via WMMA; fp32 online softmax. V arrives pre-transposed
// (Vt global layout [B,H,256,S]) so all LDS staging is vectorized.
// ---------------------------------------------------------------------------
__global__ __launch_bounds__(256) void attn_flash(
    const __bf16* __restrict__ Qb, const __bf16* __restrict__ Kb,
    const __bf16* __restrict__ Vbt, __bf16* __restrict__ attnb) {
  const int S = 2048, HD = 256;
  __shared__ __bf16 Qs[64 * 256];
  __shared__ __bf16 Ks[64 * 256];  // [key][d] == N-major for QK^T B-operand
  __shared__ __bf16 Vt[256 * 64];  // [d][key]  == N-major for PV  B-operand
  __shared__ float  Sls[64 * 64];
  __shared__ __bf16 Pls[64 * 64];
  __shared__ float  m_l[64], l_l[64], al_l[64];

  int tid  = threadIdx.x;
  int lane = tid & 31;
  int w    = tid >> 5;
  int hf   = lane >> 4;
  int r    = lane & 15;

  int qt = blockIdx.x;  // 0..31
  int h  = blockIdx.y;  // 0..15
  int b  = blockIdx.z;  // 0..1
  size_t bh = (size_t)(b * 16 + h);

  int qrow = tid >> 2, qds = (tid & 3) * 64;  // staging map: 128B per thread

  // load Q tile (64x256)
  {
    const __bf16* Qg = Qb + (bh * S + qt * 64) * HD + (size_t)qrow * HD + qds;
    __bf16* Ql = &Qs[qrow * 256 + qds];
#if USE_ASYNC
    ASYNC_B128(Qg, Ql, 0);   ASYNC_B128(Qg, Ql, 16);
    ASYNC_B128(Qg, Ql, 32);  ASYNC_B128(Qg, Ql, 48);
    ASYNC_B128(Qg, Ql, 64);  ASYNC_B128(Qg, Ql, 80);
    ASYNC_B128(Qg, Ql, 96);  ASYNC_B128(Qg, Ql, 112);
#else
#pragma unroll
    for (int e = 0; e < 8; ++e)
      *(uint4*)(Ql + e * 8) = *(const uint4*)(Qg + e * 8);
#endif
  }
  if (tid < 64) { m_l[tid] = -1e30f; l_l[tid] = 0.f; }

  v8f zero = {0.f, 0.f, 0.f, 0.f, 0.f, 0.f, 0.f, 0.f};
  v8f acc[8];
#pragma unroll
  for (int jn = 0; jn < 8; ++jn) acc[jn] = zero;
  int wm = (w >> 1) * 16;  // output row tile (0..48)
  int wn = (w & 1) * 128;  // output col half (0/128)

#if USE_ASYNC
  __builtin_amdgcn_s_wait_asynccnt(0);
#endif
  __syncthreads();

  for (int j = 0; j <= qt; ++j) {
    // stage K chunk [64 keys][256 d] and V chunk transposed [256 d][64 keys]
    {
      const __bf16* Kg = Kb + (bh * S + j * 64) * HD + (size_t)qrow * HD + qds;
      __bf16* Kl = &Ks[qrow * 256 + qds];
      const __bf16* Vg = Vbt + (bh * HD + tid) * (size_t)S + j * 64;
      __bf16* Vl = &Vt[tid * 64];
#if USE_ASYNC
      ASYNC_B128(Kg, Kl, 0);   ASYNC_B128(Kg, Kl, 16);
      ASYNC_B128(Kg, Kl, 32);  ASYNC_B128(Kg, Kl, 48);
      ASYNC_B128(Kg, Kl, 64);  ASYNC_B128(Kg, Kl, 80);
      ASYNC_B128(Kg, Kl, 96);  ASYNC_B128(Kg, Kl, 112);
      ASYNC_B128(Vg, Vl, 0);   ASYNC_B128(Vg, Vl, 16);
      ASYNC_B128(Vg, Vl, 32);  ASYNC_B128(Vg, Vl, 48);
      ASYNC_B128(Vg, Vl, 64);  ASYNC_B128(Vg, Vl, 80);
      ASYNC_B128(Vg, Vl, 96);  ASYNC_B128(Vg, Vl, 112);
      __builtin_amdgcn_s_wait_asynccnt(0);
#else
#pragma unroll
      for (int e = 0; e < 8; ++e)
        *(uint4*)(Kl + e * 8) = *(const uint4*)(Kg + e * 8);
#pragma unroll
      for (int e = 0; e < 8; ++e)
        *(uint4*)(Vl + e * 8) = *(const uint4*)(Vg + e * 8);
#endif
    }
    __syncthreads();

    // scores: 16 tiles of 16x16, 2 per wave
#pragma unroll
    for (int t2 = 0; t2 < 2; ++t2) {
      int tt = 2 * w + t2;
      int mi = (tt >> 2) * 16, ni = (tt & 3) * 16;
      v8f sacc = zero;
#pragma unroll
      for (int ks = 0; ks < 8; ++ks) {
        v16bf af = frag_A(Qs, mi, 256, ks * 32);
        v16bf bf = frag_B(Ks, ni, 256, ks * 32);
        sacc = wmma_bf16(af, bf, sacc);
      }
#pragma unroll
      for (int rr = 0; rr < 8; ++rr) {
        int rl = mi + rr + hf * 8;
        int cl = ni + r;
        float vsc = sacc[rr];
        if (j * 64 + cl > qt * 64 + rl) vsc = -1e30f;  // causal mask
        Sls[rl * 64 + cl] = vsc;
      }
    }
    __syncthreads();

    // online softmax per row (packed dword stores for P)
    if (tid < 64) {
      float mo = m_l[tid];
      float mx = mo;
#pragma unroll 8
      for (int c = 0; c < 64; ++c) mx = fmaxf(mx, Sls[tid * 64 + c]);
      float alpha = __expf(mo - mx);
      float sum = 0.f;
#pragma unroll 4
      for (int c = 0; c < 64; c += 2) {
        float p0 = __expf(Sls[tid * 64 + c] - mx);
        float p1 = __expf(Sls[tid * 64 + c + 1] - mx);
        sum += p0 + p1;
        union { __bf16 h[2]; unsigned u; } pb;
        pb.h[0] = (__bf16)p0;
        pb.h[1] = (__bf16)p1;
        *(unsigned*)&Pls[tid * 64 + c] = pb.u;
      }
      l_l[tid]  = l_l[tid] * alpha + sum;
      m_l[tid]  = mx;
      al_l[tid] = alpha;
    }
    __syncthreads();

    // rescale running O, then O += P @ V
    float arow[8];
#pragma unroll
    for (int rr = 0; rr < 8; ++rr) arow[rr] = al_l[wm + rr + hf * 8];
#pragma unroll
    for (int jn = 0; jn < 8; ++jn)
#pragma unroll
      for (int rr = 0; rr < 8; ++rr) acc[jn][rr] *= arow[rr];

#pragma unroll
    for (int ks = 0; ks < 2; ++ks) {
      v16bf af = frag_A(Pls, wm, 64, ks * 32);
#pragma unroll
      for (int jn = 0; jn < 8; ++jn) {
        v16bf bf = frag_B(Vt, wn + jn * 16, 64, ks * 32);
        acc[jn] = wmma_bf16(af, bf, acc[jn]);
      }
    }
    __syncthreads();  // protect LDS for next chunk (and l_l reads below)
  }

  // normalize + scatter O as bf16 into [b*S+s][h*256+d]
  float rl_[8];
#pragma unroll
  for (int rr = 0; rr < 8; ++rr) rl_[rr] = 1.0f / l_l[wm + rr + hf * 8];
#pragma unroll
  for (int jn = 0; jn < 8; ++jn)
#pragma unroll
    for (int rr = 0; rr < 8; ++rr) {
      int rowl = wm + rr + hf * 8;
      int col  = wn + jn * 16 + r;
      size_t srow = (size_t)b * 2048 + qt * 64 + rowl;
      attnb[srow * 4096 + h * 256 + col] = (__bf16)(acc[jn][rr] * rl_[rr]);
    }
}

// ---------------------------------------------------------------------------
// Host launcher
// ---------------------------------------------------------------------------
extern "C" void kernel_launch(void* const* d_in, const int* in_sizes, int n_in,
                              void* d_out, int out_size, void* d_ws,
                              size_t ws_size, hipStream_t stream) {
  const float* hidden = (const float*)d_in[0];  // [2,2048,4096] fp32
  const int*   pos    = (const int*)d_in[1];    // [2,2048] int32
  const float* Wqkv   = (const float*)d_in[2];  // [4096,12288] fp32
  const float* Wout   = (const float*)d_in[3];  // [4096,4096] fp32
  float* out = (float*)d_out;                   // [2,2048,4096] fp32

  const size_t BS = 4096;  // B*S
  const size_t D  = 4096;
  const size_t D3 = 12288;
  const size_t HEAD_ELEMS = (size_t)2 * 16 * 2048 * 256;

  char* ws = (char*)d_ws;
  size_t off = 0;
  __bf16* hid_b  = (__bf16*)(ws + off); off += BS * D  * 2;
  __bf16* wqkvT  = (__bf16*)(ws + off); off += D  * D3 * 2;  // [12288][4096]
  __bf16* woutT  = (__bf16*)(ws + off); off += D  * D  * 2;  // [4096][4096]
  float*  qkv    = (float*)(ws + off);  off += BS * D3 * 4;
  __bf16* Qb     = (__bf16*)(ws + off); off += HEAD_ELEMS * 2;
  __bf16* Kb     = (__bf16*)(ws + off); off += HEAD_ELEMS * 2;
  __bf16* Vb     = (__bf16*)(ws + off); off += HEAD_ELEMS * 2;
  __bf16* Vbt    = (__bf16*)(ws + off); off += HEAD_ELEMS * 2;  // [B,H,256,S]
  __bf16* attnb  = (__bf16*)(ws + off); off += BS * D * 2;

  // 1) convert A-operand; convert+transpose weights to N-major bf16
  cvt_f32_bf16<<<4096, 256, 0, stream>>>(hidden, hid_b, BS * D);
  cvt_transpose_f32_bf16<<<dim3(12288 / 64, 4096 / 64), 256, 0, stream>>>(
      Wqkv, wqkvT, 4096, 12288);
  cvt_transpose_f32_bf16<<<dim3(4096 / 64, 4096 / 64), 256, 0, stream>>>(
      Wout, woutT, 4096, 4096);

  // 2) qkv = hidden @ W_qkv   (4096 x 12288 x 4096)
  gemm_bf16_f32<12288, 4096><<<dim3(12288 / GT_N, 4096 / GT_M), 256, 0,
                               stream>>>(hid_b, wqkvT, qkv);

  // 3) rotary + scatter per-head Q/K/V (q pre-scaled by 1/sqrt(256))
  rotary_scatter<<<dim3(16, 2048, 2), 128, 0, stream>>>(qkv, pos, Qb, Kb, Vb);

  // 4) one-time V transpose to d-major [B,H,256,2048]
  transpose_bf16<<<dim3(256 / 64, 2048 / 64, 32), 256, 0, stream>>>(
      Vb, Vbt, 2048, 256);

  // 5) flash attention per (b, h, 64-row query tile)
  attn_flash<<<dim3(32, 16, 2), 256, 0, stream>>>(Qb, Kb, Vbt, attnb);

  // 6) out = attn @ W_out   (4096 x 4096 x 4096)
  gemm_bf16_f32<4096, 4096><<<dim3(4096 / GT_N, 4096 / GT_M), 256, 0,
                              stream>>>(attnb, woutT, out);
}
